// SynthesisLayer_2542620639936
// MI455X (gfx1250) — matrix operand order repaired and checked by reading, compiled, not run on hardware
//
#include <hip/hip_runtime.h>
#include <math.h>

// ---------------- types / helpers ----------------
typedef __attribute__((ext_vector_type(8)))  float        v8f;
typedef __attribute__((ext_vector_type(16))) __bf16       bf16x16;
typedef __attribute__((ext_vector_type(8)))  __bf16       bf16x8;
typedef __attribute__((ext_vector_type(4)))  unsigned int v4u;
typedef __attribute__((ext_vector_type(8)))  int          v8i;
typedef __attribute__((ext_vector_type(4)))  int          v4i;

union U16B { uint4 u; bf16x8 v; };

__device__ __forceinline__ unsigned short f2bf(float f) {
  union { float f; unsigned u; } v; v.f = f;
  unsigned r = v.u + 0x7FFFu + ((v.u >> 16) & 1u);
  return (unsigned short)(r >> 16);
}
__device__ __forceinline__ float bf2f(unsigned short h) {
  union { unsigned u; float f; } v; v.u = ((unsigned)h) << 16;
  return v.f;
}

#define CI 512
#define CO 512
#define HO 66          // conv output 66x66
#define MTOT (66*66)   // 4356
#define YSTR 4360      // padded channel stride (16B aligned in bf16)
#define HP 68          // padded input 68x68
#define KITERS 144     // 9 taps * 16 k-blocks (even -> clean 2x unroll)

#if defined(__gfx1250__) && __has_builtin(__builtin_amdgcn_tensor_load_to_lds) && __has_builtin(__builtin_amdgcn_s_wait_tensorcnt)
#define HAVE_TDM 1
#else
#define HAVE_TDM 0
#endif

#if HAVE_TDM
// 2D TDM descriptor: 8 rows x 1024B, global row stride 16384B, data_size=4B.
// 6-arg builtin form: (v4u, v8i, v4i, v4i, v8i, i32 cpol)
__device__ __forceinline__ void tdm_load_2d(const void* gptr, unsigned lds_off) {
  unsigned long long ga = (unsigned long long)(size_t)gptr;
  v4u g0;
  g0.x = 1u;                                            // count=1 (valid user D#)
  g0.y = lds_off;                                       // LDS byte address
  g0.z = (unsigned)ga;                                  // global_addr[31:0]
  g0.w = (unsigned)((ga >> 32) & 0x01FFFFFFu) | 0x80000000u; // addr[56:32] | type=2
  v8i g1 = (v8i){ 0x00020000,   // wg_mask=0, data_size=4B
                  0x10000000,   // tensor_dim0 lo16 (=4096) << 16
                  0x00400000,   // tensor_dim0 hi16=0 | tensor_dim1 lo16 (=64) << 16
                  0x01000000,   // tensor_dim1 hi16=0 | tile_dim0 (=256) << 16
                  8,            // tile_dim1=8, tile_dim2=0
                  4096,         // tensor_dim0_stride lo32 (units of 4B) = 16384B
                  0, 0 };
  v4i gz  = (v4i){0, 0, 0, 0};
  v8i gz8 = (v8i){0, 0, 0, 0, 0, 0, 0, 0};
  __builtin_amdgcn_tensor_load_to_lds(g0, g1, gz, gz, gz8, 0);
}
#endif

// ---------------- kernel 1a: styles = w @ A^T / sqrt(512) + bias ----------------
__global__ __launch_bounds__(256) void k_styles(const float* __restrict__ w,
                                                const float* __restrict__ aw,
                                                const float* __restrict__ ab,
                                                float* __restrict__ s_raw) {
  int gid = blockIdx.x * 256 + threadIdx.x;   // 4096
  int n = gid >> 9, c = gid & 511;
  const float* wr = w + n * 512;
  const float* ar = aw + (size_t)c * 512;
  float acc = 0.f;
  for (int k = 0; k < 512; ++k) acc += wr[k] * ar[k];
  s_raw[gid] = acc * 0.04419417382415922f + ab[c];   // 1/sqrt(512)
}

// ---------------- kernel 1b: weight norm, demod d[n,o], WMMA-B layout ----------------
// wB element index: (((t*32 + o/16)*16 + kb)*32 + lane)*16 + j
__global__ __launch_bounds__(256) void k_wprep(const float* __restrict__ cw,
                                               const float* __restrict__ s_raw,
                                               const float* __restrict__ mag,
                                               unsigned short* __restrict__ wB,
                                               float* __restrict__ d) {
  __shared__ float wsh[4608];
  __shared__ float qsh[512];
  __shared__ float red[256];
  int o = blockIdx.x, tid = threadIdx.x;

  float ss = 0.f;
  for (int idx = tid; idx < 4608; idx += 256) {
    float v = cw[(size_t)o * 4608 + idx];
    wsh[idx] = v; ss += v * v;
  }
  red[tid] = ss; __syncthreads();
  for (int off = 128; off > 0; off >>= 1) { if (tid < off) red[tid] += red[tid + off]; __syncthreads(); }
  float rn = rsqrtf(red[0] / 4608.0f);
  __syncthreads();
  for (int idx = tid; idx < 4608; idx += 256) wsh[idx] *= rn;

  float s2 = 0.f;
  for (int idx = tid; idx < 4096; idx += 256) { float v = s_raw[idx]; s2 += v * v; }
  __syncthreads();
  red[tid] = s2; __syncthreads();
  for (int off = 128; off > 0; off >>= 1) { if (tid < off) red[tid] += red[tid + off]; __syncthreads(); }
  float rs = rsqrtf(red[0] / 4096.0f);
  __syncthreads();

  for (int i = tid; i < 512; i += 256) {
    float q = 0.f;
    #pragma unroll
    for (int k = 0; k < 9; ++k) { float v = wsh[i * 9 + k]; q += v * v; }
    qsh[i] = q;
  }
  __syncthreads();

  float ig = rsqrtf(mag[0]);
  float part[8];
  #pragma unroll
  for (int n = 0; n < 8; ++n) part[n] = 0.f;
  for (int i = tid; i < 512; i += 256) {
    float q = qsh[i];
    #pragma unroll
    for (int n = 0; n < 8; ++n) { float sv = s_raw[n * 512 + i] * rs; part[n] += q * sv * sv; }
  }
  for (int n = 0; n < 8; ++n) {
    __syncthreads();
    red[tid] = part[n]; __syncthreads();
    for (int off = 128; off > 0; off >>= 1) { if (tid < off) red[tid] += red[tid + off]; __syncthreads(); }
    if (tid == 0) d[n * 512 + o] = rsqrtf(red[0] + 1e-8f) * ig;
  }
  __syncthreads();

  for (int idx = tid; idx < 4608; idx += 256) {
    int j = idx & 15; int r = idx >> 4;
    int Lhi = r & 1; r >>= 1;
    int kb = r & 15; int t = r >> 4;
    int i = kb * 32 + ((j & 8) << 1) + Lhi * 8 + (j & 7);
    int lane = (o & 15) + Lhi * 16;
    size_t off = ((((size_t)t * 32 + (o >> 4)) * 16 + kb) * 32 + lane) * 16 + j;
    wB[off] = f2bf(wsh[i * 9 + t]);
  }
}

// ---------------- kernel 2: x_mod = bf16( x * s_norm ), padded NHWC ----------------
__global__ __launch_bounds__(256) void k_xmod(const float* __restrict__ x,
                                              const float* __restrict__ s_raw,
                                              unsigned short* __restrict__ xp) {
  __shared__ float tile[64][65];
  __shared__ float red[256];
  int tid = threadIdx.x;
  int py = blockIdx.x;   // 0..67
  int n  = blockIdx.y;   // 0..7

  float s2 = 0.f;
  for (int idx = tid; idx < 4096; idx += 256) { float v = s_raw[idx]; s2 += v * v; }
  red[tid] = s2; __syncthreads();
  for (int off = 128; off > 0; off >>= 1) { if (tid < off) red[tid] += red[tid + off]; __syncthreads(); }
  float rs = rsqrtf(red[0] / 4096.0f);

  size_t rowbase = (((size_t)n * HP) + py) * HP * CI;
  int ysrc = py - 2;
  if (ysrc < 0 || ysrc > 63) {
    for (int idx = tid; idx < HP * CI; idx += 256) xp[rowbase + idx] = 0;
    return;
  }
  for (int idx = tid; idx < 4 * CI; idx += 256) {
    int c = idx >> 9, i = idx & 511;
    int px = (c < 2) ? c : (64 + c);
    xp[rowbase + (size_t)px * CI + i] = 0;
  }
  for (int ic = 0; ic < CI; ic += 64) {
    __syncthreads();
    #pragma unroll
    for (int rep = 0; rep < 16; ++rep) {
      int flat = rep * 256 + tid;
      int ii = flat >> 6, xx = flat & 63;
      tile[ii][xx] = x[((((size_t)n * CI) + ic + ii) * 64 + ysrc) * 64 + xx];
    }
    __syncthreads();
    #pragma unroll
    for (int rep = 0; rep < 16; ++rep) {
      int flat = rep * 256 + tid;
      int xx = flat >> 6, ii = flat & 63;
      float sv = s_raw[n * 512 + ic + ii] * rs;
      xp[rowbase + (size_t)(xx + 2) * CI + (ic + ii)] = f2bf(tile[ii][xx] * sv);
    }
  }
}

// ---------------- kernel 3: implicit-GEMM conv via WMMA bf16 + TDM B staging ----------------
// Block: 128(M) x 128(N); 8 waves (4M x 2N); wave tile 2x4 16x16.
// K loop flattened (it = t*16 + kb), manually unrolled x2 so the ping-pong
// register buffers are statically indexed (no scratch spills). All 8 B-fragment
// ds_loads are issued before the WMMA block so LDS latency overlaps WMMA issue.
__device__ __forceinline__ void gemm_step(
    int it, bool hasnext, int wv,
    const unsigned short* __restrict__ wB, unsigned oby, unsigned lds_nxt,
    const unsigned short* a0, const unsigned short* a1,
    U16B* alo_c, U16B* ahi_c, U16B* alo_n, U16B* ahi_n,
    const unsigned short* bbase_c, int nfstride,
    v8f (&acc)[2][4]) {
#if HAVE_TDM
  __syncthreads();                       // all waves done reading next buffer's old data
  if (wv == 0) {
    if (hasnext) {
      int t2 = (it + 1) >> 4, kb2 = (it + 1) & 15;
      size_t e = ((((size_t)t2 * 32 + oby) * 16 + kb2) * 32) * 16;
      tdm_load_2d((const void*)(wB + e), lds_nxt);
      __builtin_amdgcn_s_wait_tensorcnt(1);   // current buffer's DMA complete
    } else {
      __builtin_amdgcn_s_wait_tensorcnt(0);
    }
  }
  __syncthreads();                       // publish current buffer
#endif
  // issue all B-fragment loads up front (staggered dscnt waits at the WMMAs)
  U16B blo[4], bhi[4];
  #pragma unroll
  for (int nf = 0; nf < 4; ++nf) {
    const unsigned short* q = bbase_c + (size_t)nf * nfstride;
    blo[nf].u = *(const uint4*)(q);
    bhi[nf].u = *(const uint4*)(q + 8);
  }
  // A prefetch for next step (global) overlaps with this step's WMMAs
  if (hasnext) {
    int t2 = (it + 1) >> 4, kb2 = (it + 1) & 15;
    int ky2 = t2 / 3, kx2 = t2 - ky2 * 3;
    size_t aoff = ((size_t)ky2 * HP + kx2) * CI + kb2 * 32;
    const unsigned short* p0 = a0 + aoff;
    const unsigned short* p1 = a1 + aoff;
    alo_n[0].u = *(const uint4*)(p0);
    ahi_n[0].u = *(const uint4*)(p0 + 16);
    alo_n[1].u = *(const uint4*)(p1);
    ahi_n[1].u = *(const uint4*)(p1 + 16);
    if ((it & 15) == 0)
      __builtin_prefetch((const void*)(p0 + 512), 0, 1);
  }
  bf16x16 afrag[2];
  #pragma unroll
  for (int mf = 0; mf < 2; ++mf) {
    #pragma unroll
    for (int j = 0; j < 8; ++j) {
      afrag[mf][j]     = alo_c[mf].v[j];
      afrag[mf][8 + j] = ahi_c[mf].v[j];
    }
  }
  #pragma unroll
  for (int nf = 0; nf < 4; ++nf) {
    bf16x16 bfrag;
    #pragma unroll
    for (int j = 0; j < 8; ++j) { bfrag[j] = blo[nf].v[j]; bfrag[8 + j] = bhi[nf].v[j]; }
    acc[0][nf] = __builtin_amdgcn_wmma_f32_16x16x32_bf16(
        false, afrag[0], false, bfrag, (short)0, acc[0][nf], false, false);
    acc[1][nf] = __builtin_amdgcn_wmma_f32_16x16x32_bf16(
        false, afrag[1], false, bfrag, (short)0, acc[1][nf], false, false);
  }
}

__global__ __launch_bounds__(256) void k_conv_gemm(const unsigned short* __restrict__ xp,
                                                   const unsigned short* __restrict__ wB,
                                                   const float* __restrict__ d,
                                                   unsigned short* __restrict__ yout) {
#if HAVE_TDM
  __shared__ __align__(16) unsigned short Bsh[2][4096];   // 2 x 8KB
#endif
  int tid  = threadIdx.x;
  int lane = tid & 31, wv = tid >> 5;
  int wm = wv & 3, wn = wv >> 2;
  int n      = blockIdx.z;
  int m_blk  = blockIdx.x * 128 + wm * 32;
  int o_base = blockIdx.y * 128 + wn * 64;
  unsigned oby = blockIdx.y * 8;
  const unsigned short* xpn = xp + (size_t)n * HP * HP * CI;

  int lrow = lane & 15;
  int klo  = (lane < 16) ? 0 : 8;
  const unsigned short* aptr0;
  const unsigned short* aptr1;
  {
    int m0 = m_blk + lrow;        if (m0 > MTOT - 1) m0 = MTOT - 1;
    int m1 = m_blk + 16 + lrow;   if (m1 > MTOT - 1) m1 = MTOT - 1;
    aptr0 = xpn + ((size_t)(m0 / HO) * HP + (m0 % HO)) * CI + klo;
    aptr1 = xpn + ((size_t)(m1 / HO) * HP + (m1 % HO)) * CI + klo;
  }

  v8f acc[2][4];
  #pragma unroll
  for (int mf = 0; mf < 2; ++mf)
    #pragma unroll
    for (int nf = 0; nf < 4; ++nf) acc[mf][nf] = (v8f){0, 0, 0, 0, 0, 0, 0, 0};

  // statically-named A ping-pong buffers (registers)
  U16B alo0[2], ahi0[2], alo1[2], ahi1[2];
  alo0[0].u = *(const uint4*)(aptr0);
  ahi0[0].u = *(const uint4*)(aptr0 + 16);
  alo0[1].u = *(const uint4*)(aptr1);
  ahi0[1].u = *(const uint4*)(aptr1 + 16);

#if HAVE_TDM
  unsigned lds0 = (unsigned)(size_t)(&Bsh[0][0]);
  unsigned lds1 = (unsigned)(size_t)(&Bsh[1][0]);
  const unsigned short* bb0 = &Bsh[0][((size_t)(wn * 4)) * 512 + (size_t)lane * 16];
  const unsigned short* bb1 = &Bsh[1][((size_t)(wn * 4)) * 512 + (size_t)lane * 16];
  if (wv == 0) {
    size_t e0 = ((size_t)oby * 16) * 32 * 16;   // it=0 -> t=0, kb=0
    tdm_load_2d((const void*)(wB + e0), lds0);
  }
  for (int itp = 0; itp < KITERS; itp += 2) {
    gemm_step(itp,     true,               wv, wB, oby, lds1, aptr0, aptr1,
              alo0, ahi0, alo1, ahi1, bb0, 512, acc);
    gemm_step(itp + 1, itp + 2 < KITERS,   wv, wB, oby, lds0, aptr0, aptr1,
              alo1, ahi1, alo0, ahi0, bb1, 512, acc);
  }
#else
  for (int itp = 0; itp < KITERS; itp += 2) {
    {
      int t = itp >> 4, kb = itp & 15;
      const unsigned short* bb = wB +
          ((((size_t)t * 32 + (o_base >> 4)) * 16 + kb) * 32 + lane) * 16;
      gemm_step(itp, true, wv, wB, oby, 0, aptr0, aptr1,
                alo0, ahi0, alo1, ahi1, bb, 8192, acc);
    }
    {
      int it = itp + 1;
      int t = it >> 4, kb = it & 15;
      const unsigned short* bb = wB +
          ((((size_t)t * 32 + (o_base >> 4)) * 16 + kb) * 32 + lane) * 16;
      gemm_step(it, it + 1 < KITERS, wv, wB, oby, 0, aptr0, aptr1,
                alo1, ahi1, alo0, ahi0, bb, 8192, acc);
    }
  }
#endif

  // epilogue: demod scale, bf16 store (lane holds 8 contiguous spatial M)
  #pragma unroll
  for (int nf = 0; nf < 4; ++nf) {
    int o = o_base + nf * 16 + lrow;
    float dd = d[n * 512 + o];
    #pragma unroll
    for (int mf = 0; mf < 2; ++mf) {
      int mstart = m_blk + mf * 16 + ((lane < 16) ? 0 : 8);
      if (mstart > MTOT - 1) continue;
      unsigned short* yp = yout + (size_t)(n * 512 + o) * YSTR + mstart;
      v8f c = acc[mf][nf];
      if (mstart + 7 <= MTOT - 1) {
        unsigned pk[4];
        #pragma unroll
        for (int j = 0; j < 4; ++j) {
          unsigned a0 = f2bf(c[2 * j] * dd);
          unsigned a1 = f2bf(c[2 * j + 1] * dd);
          pk[j] = a0 | (a1 << 16);
        }
        *(uint4*)yp = make_uint4(pk[0], pk[1], pk[2], pk[3]);
      } else {
        int nv = MTOT - mstart;
        for (int j = 0; j < 8; ++j) if (j < nv) yp[j] = f2bf(c[j] * dd);
      }
    }
  }
}

// ---------------- kernel 4: fused bias -> up2 FIR -> lrelu/clamp -> down2 FIR ----------------
__global__ __launch_bounds__(256) void k_flrelu(const unsigned short* __restrict__ yin,
                                                const float* __restrict__ cb,
                                                const float* __restrict__ fup,
                                                const float* __restrict__ fdn,
                                                float* __restrict__ out) {
  __shared__ float t1[138 * 66];
  __shared__ float zs[26 * 140];
  __shared__ float dv[8 * 140];
  int ch = blockIdx.x, tid = threadIdx.x;
  int o = ch & 511;
  float bias = cb[o];
  float fru[12], frd[12];
  #pragma unroll
  for (int t = 0; t < 12; ++t) { fru[t] = fup[11 - t]; frd[t] = fdn[11 - t]; }
  const unsigned short* y = yin + (size_t)ch * YSTR;

  for (int idx = tid; idx < 138 * 66; idx += 256) {
    int u = idx / 66, xx = idx - u * 66;
    float acc = 0.f;
    for (int t = ((u + 1) & 1); t < 12; t += 2) {
      int s = u + t - 9;
      if (s >= 0 && s <= 130) acc += fru[t] * (bf2f(y[(s >> 1) * 66 + xx]) + bias);
    }
    t1[idx] = acc;
  }
  __syncthreads();

  for (int c4 = 0; c4 < 8; ++c4) {
    int oy0 = c4 * 8;
    for (int idx = tid; idx < 26 * 138; idx += 256) {
      int uu = idx / 138, v = idx - uu * 138;
      int u = 2 * oy0 + uu;
      float acc = 0.f;
      for (int t = ((v + 1) & 1); t < 12; t += 2) {
        int s = v + t - 9;
        if (s >= 0 && s <= 130) acc += fru[t] * t1[u * 66 + (s >> 1)];
      }
      acc *= 4.0f;
      acc = (acc >= 0.f ? acc : 0.2f * acc) * 1.4142135623730951f;
      acc = fminf(fmaxf(acc, -256.f), 256.f);
      zs[uu * 140 + v] = acc;
    }
    __syncthreads();
    for (int idx = tid; idx < 8 * 138; idx += 256) {
      int r = idx / 138, v = idx - r * 138;
      float acc = 0.f;
      #pragma unroll
      for (int ta = 0; ta < 12; ++ta) acc += frd[ta] * zs[(2 * r + ta) * 140 + v];
      dv[r * 140 + v] = acc;
    }
    __syncthreads();
    for (int idx = tid; idx < 8 * 64; idx += 256) {
      int r = idx >> 6, ox = idx & 63;
      float acc = 0.f;
      #pragma unroll
      for (int tb = 0; tb < 12; ++tb) acc += frd[tb] * dv[r * 140 + 2 * ox + tb];
      out[((size_t)ch * 64 + (oy0 + r)) * 64 + ox] = acc;
    }
    __syncthreads();
  }
}

// ---------------- launch ----------------
extern "C" void kernel_launch(void* const* d_in, const int* in_sizes, int n_in,
                              void* d_out, int out_size, void* d_ws, size_t ws_size,
                              hipStream_t stream) {
  const float* x   = (const float*)d_in[0];
  const float* w   = (const float*)d_in[1];
  const float* aw  = (const float*)d_in[2];
  const float* ab  = (const float*)d_in[3];
  const float* cw  = (const float*)d_in[4];
  const float* cb  = (const float*)d_in[5];
  const float* fup = (const float*)d_in[6];
  const float* fdn = (const float*)d_in[7];
  const float* mag = (const float*)d_in[8];

  char* ws = (char*)d_ws;
  float*          s_raw = (float*)(ws);
  float*          dbuf  = (float*)(ws + 32768);
  unsigned short* wB    = (unsigned short*)(ws + 65536);
  unsigned short* xp    = (unsigned short*)(ws + ((size_t)8  << 20));
  unsigned short* yb    = (unsigned short*)(ws + ((size_t)48 << 20));
  float* out = (float*)d_out;

  k_styles   <<<16, 256, 0, stream>>>(w, aw, ab, s_raw);
  k_wprep    <<<512, 256, 0, stream>>>(cw, s_raw, mag, wB, dbuf);
  k_xmod     <<<dim3(68, 8), 256, 0, stream>>>(x, s_raw, xp);
  k_conv_gemm<<<dim3(35, 4, 8), 256, 0, stream>>>(xp, wB, dbuf, yb);
  k_flrelu   <<<4096, 256, 0, stream>>>(yb, cb, fup, fdn, out);
}
